// ConvNN_2D_Spatial_K_N_Location_20435454394593
// MI455X (gfx1250) — matrix operand order repaired and checked by reading, not compiled, not run on hardware
//
#include <hip/hip_runtime.h>
#include <hip/hip_bf16.h>

// ---------- CDNA5 WMMA types ----------
typedef __attribute__((ext_vector_type(16))) __bf16        v16bf;
typedef __attribute__((ext_vector_type(8)))  float         v8f;
typedef __attribute__((ext_vector_type(8)))  unsigned int  v8u;
typedef __attribute__((ext_vector_type(4)))  unsigned int  v4u;

union FragU { v8u u; v16bf bf; v4u u4[2]; unsigned int s[8]; };

__device__ __forceinline__ unsigned short bfbits(float x) {
    unsigned int u = __builtin_bit_cast(unsigned int, x);
    u += 0x7FFFu + ((u >> 16) & 1u);           // round-to-nearest-even
    return (unsigned short)(u >> 16);
}
__device__ __forceinline__ unsigned int bfpack(float lo, float hi) {
    return (unsigned int)bfbits(lo) | ((unsigned int)bfbits(hi) << 16);
}

// candidate grid index: int32(linspace(0,15,8)) = {0,2,4,6,8,10,12,15}
__device__ __forceinline__ int ihv(int a) { return (a == 7) ? 15 : (a << 1); }

// CDNA5 async global->LDS copy (ASYNCcnt path), saddr + 32-bit voffset form.
__device__ __forceinline__ void async_ld_b128(unsigned lds_off, unsigned gvoff,
                                              const void* sbase) {
    asm volatile("global_load_async_to_lds_b128 %0, %1, %2"
                 :: "v"(lds_off), "v"(gvoff), "s"(sbase) : "memory");
}
__device__ __forceinline__ void wait_async0() {
    asm volatile("s_wait_asynccnt 0" ::: "memory");
}
__device__ __forceinline__ void wait_async4() {   // allow 4 in flight (next chunk)
    asm volatile("s_wait_asynccnt 4" ::: "memory");
}
__device__ __forceinline__ unsigned lds_off32(const void* p) {
    return (unsigned)(size_t)p;   // low 32 bits of generic LDS addr = wave LDS offset
}

// =====================================================================
// Kernel 1: layer-1 ConvNN (14 chan, 64 cand, top-9, 64 outputs) — VALU
// one block per image, one thread per token; stores PRE-shuffle layout
// h1[b][o][t] (t = h16*16+w16), which is exactly layer-2's input.
// =====================================================================
__global__ __launch_bounds__(256) void layer1_knn(
    const float* __restrict__ x,   // [1024][3][32][32]
    const float* __restrict__ w1,  // [64][14][9]
    const float* __restrict__ b1,  // [64]
    float* __restrict__ h1)        // [1024][64][256]
{
    __shared__ float samp[14][64];
    __shared__ float wsh[64 * 14 * 9];
    const int b = blockIdx.x;
    const int tid = threadIdx.x;

    for (int e = tid; e < 64 * 14 * 9; e += 256) wsh[e] = w1[e];
    for (int e = tid; e < 14 * 64; e += 256) {
        int c = e >> 6, q = e & 63;
        int hh = ihv(q >> 3), ww = ihv(q & 7);
        float v;
        if (c < 12) {
            int cc = c >> 2, ii = (c >> 1) & 1, jj = c & 1;
            v = x[((size_t)b * 3 + cc) * 1024 + (size_t)(2 * hh + ii) * 32 + (2 * ww + jj)];
        } else if (c == 12) v = hh * (1.0f / 15.0f);
        else                v = ww * (1.0f / 15.0f);
        samp[c][q] = v;
    }
    __syncthreads();

    const int t = tid, th = t >> 4, tw = t & 15;
    float f[14];
#pragma unroll
    for (int c = 0; c < 12; ++c) {
        int cc = c >> 2, ii = (c >> 1) & 1, jj = c & 1;
        f[c] = x[((size_t)b * 3 + cc) * 1024 + (size_t)(2 * th + ii) * 32 + (2 * tw + jj)];
    }
    f[12] = th * (1.0f / 15.0f);
    f[13] = tw * (1.0f / 15.0f);

    float d2[64];
#pragma unroll
    for (int q = 0; q < 64; ++q) {
        float s = 0.f;
#pragma unroll
        for (int c = 0; c < 14; ++c) { float d = f[c] - samp[c][q]; s += d * d; }
        d2[q] = s;
    }

    // stable top-9 (smallest d2, lowest index wins ties — matches lax.top_k)
    unsigned long long used = 0ull;
    int sel[9];
#pragma unroll
    for (int k = 0; k < 9; ++k) {
        float best = 3.4e38f; int bi = 0;
#pragma unroll
        for (int q = 0; q < 64; ++q) {
            bool ok = !((used >> q) & 1ull);
            if (ok && d2[q] < best) { best = d2[q]; bi = q; }
        }
        used |= (1ull << bi);
        sel[k] = bi;
    }

    float ng[9][14];
#pragma unroll
    for (int k = 0; k < 9; ++k)
#pragma unroll
        for (int c = 0; c < 14; ++c) ng[k][c] = samp[c][sel[k]];

    for (int o = 0; o < 64; ++o) {
        float acc = b1[o];
#pragma unroll
        for (int k = 0; k < 9; ++k)
#pragma unroll
            for (int c = 0; c < 14; ++c) acc += ng[k][c] * wsh[(o * 14 + c) * 9 + k];
        h1[((size_t)b * 64 + o) * 256 + t] = acc > 0.f ? acc : 0.f;
    }
}

// =====================================================================
// Kernel 2: layer-2 fused KNN + bf16 WMMA GEMM per image.
// C(128x256) = W2(128x608pad) x Neigh(608pad x 256), f32 accumulate.
// B chunk stored as packed bf16 pairs -> b128 fragment loads.
// W2 chunk staged via CDNA5 async global->LDS, overlapped with B build.
// =====================================================================
__global__ __launch_bounds__(256) void layer2_knn_wmma(
    const float* __restrict__ h1,            // [1024][64][256]
    const unsigned short* __restrict__ w2bf, // [128][608] bf16 (594 valid)
    const float* __restrict__ b2,            // [128]
    unsigned short* __restrict__ h2bf)       // [1024][32768] bf16 (shuffled)
{
    __shared__ __align__(16) float        samp[66][64];   // 16.5 KB
    __shared__ __align__(16) int          nidx[256][9];   //  9.0 KB
    __shared__ __align__(16) unsigned int nbufU[256][16]; // 16.0 KB  B: [n][kpair]
    __shared__ __align__(16) unsigned int wbufU[128][16]; //  8.0 KB  A: [m][kpair]
    const int b = blockIdx.x;
    const int tid = threadIdx.x;
    const int wave = tid >> 5, lane = tid & 31;

    // candidates (64 data ch + 2 loc ch)
    for (int e = tid; e < 66 * 64; e += 256) {
        int c = e >> 6, q = e & 63;
        int hh = ihv(q >> 3), ww = ihv(q & 7);
        float v;
        if (c < 64)       v = h1[((size_t)b * 64 + c) * 256 + hh * 16 + ww];
        else if (c == 64) v = hh * (1.0f / 15.0f);
        else              v = ww * (1.0f / 15.0f);
        samp[c][q] = v;
    }
    __syncthreads();

    // per-token KNN (66-dim)
    {
        const int t = tid;
        float f[66];
#pragma unroll
        for (int c = 0; c < 64; ++c) f[c] = h1[((size_t)b * 64 + c) * 256 + t];
        f[64] = (t >> 4) * (1.0f / 15.0f);
        f[65] = (t & 15) * (1.0f / 15.0f);
        float d2[64];
#pragma unroll
        for (int q = 0; q < 64; ++q) {
            float s = 0.f;
#pragma unroll
            for (int c = 0; c < 66; ++c) { float d = f[c] - samp[c][q]; s += d * d; }
            d2[q] = s;
        }
        unsigned long long used = 0ull;
#pragma unroll
        for (int k = 0; k < 9; ++k) {
            float best = 3.4e38f; int bi = 0;
#pragma unroll
            for (int q = 0; q < 64; ++q) {
                bool ok = !((used >> q) & 1ull);
                if (ok && d2[q] < best) { best = d2[q]; bi = q; }
            }
            used |= (1ull << bi);
            nidx[t][k] = bi;
        }
    }
    __syncthreads();

    // GEMM: wave w owns M rows [16w,16w+16) x all 256 cols -> 16 tiles
    v8f acc[16];
#pragma unroll
    for (int nt = 0; nt < 16; ++nt)
#pragma unroll
        for (int j = 0; j < 8; ++j) acc[nt][j] = 0.f;

    const int t = tid;
    const unsigned ldsW = lds_off32(&wbufU[0][0]);
    const int rr  = tid >> 1;            // w2 row staged by this thread
    const int cbb = (tid & 1) * 8;       // uint col base within 16-uint row

    for (int k0 = 0; k0 < 608; k0 += 32) {
        // stage A chunk (w2 rows) via async global->LDS (2 x b128 per thread)
        {
            unsigned gb = (unsigned)(rr * 1216 + k0 * 2 + cbb * 4);
            unsigned lb = ldsW + (unsigned)((rr * 16 + cbb) * 4);
            async_ld_b128(lb,      gb,      (const void*)w2bf);
            async_ld_b128(lb + 16, gb + 16, (const void*)w2bf);
        }
        // build B chunk as packed pairs: nbufU[t][p] = bf16(k0+2p) | bf16(k0+2p+1)<<16
        // (long VALU/LDS section -> async weight load latency is hidden here)
#pragma unroll
        for (int p = 0; p < 16; ++p) {
            int r0 = k0 + 2 * p;
            int r1 = r0 + 1;
            float v0 = 0.f, v1 = 0.f;
            if (r0 < 594) { int c = r0 / 9, ks = r0 - c * 9; v0 = samp[c][nidx[t][ks]]; }
            if (r1 < 594) { int c = r1 / 9, ks = r1 - c * 9; v1 = samp[c][nidx[t][ks]]; }
            nbufU[t][p] = bfpack(v0, v1);
        }
        wait_async0();
        __syncthreads();

        // A frag: 16x32 bf16 (lanes16-31 K-base 8; VGPR4-7 hold K+16) -> 2 x b128
        const int am = wave * 16 + (lane & 15);
        const int a0 = (lane >= 16) ? 1 : 0;
        const v4u* pA = (const v4u*)&wbufU[am][0];
        FragU af;
        af.u4[0] = pA[a0];
        af.u4[1] = pA[a0 + 2];
        const int i0 = (lane >= 16) ? 1 : 0;   // B K-base 8 for upper half-wave; second half holds K+16
#pragma unroll
        for (int nt = 0; nt < 16; ++nt) {
            const int bn = nt * 16 + (lane & 15);
            const v4u* pB = (const v4u*)&nbufU[bn][0];
            FragU bfr;
            bfr.u4[0] = pB[i0];
            bfr.u4[1] = pB[i0 + 2];
            acc[nt] = __builtin_amdgcn_wmma_f32_16x16x32_bf16(
                false, af.bf, false, bfr.bf, (short)0, acc[nt], false, false);
        }
        __syncthreads();
    }

    // epilogue: bias + ReLU + bf16, pixel-shuffled flat store
    const int mbase = wave * 16 + ((lane >= 16) ? 8 : 0);
#pragma unroll
    for (int nt = 0; nt < 16; ++nt) {
        const int tc = nt * 16 + (lane & 15);
        const int h16 = tc >> 4, w16 = tc & 15;
#pragma unroll
        for (int i = 0; i < 8; ++i) {
            const int o = mbase + i;
            float v = acc[nt][i] + b2[o];
            v = v > 0.f ? v : 0.f;
            const int cch = o >> 2, ii = (o >> 1) & 1, jj = o & 1;
            const size_t flat = (size_t)cch * 1024 + (size_t)(2 * h16 + ii) * 32 + (2 * w16 + jj);
            h2bf[(size_t)b * 32768 + flat] = bfbits(v);
        }
    }
}

// =====================================================================
// Kernel 3: FC1 — bf16 WMMA GEMM, C(1024x1024)=H(1024x32768)*W^T,
// 128x128 tile per block, DOUBLE-BUFFERED async global->LDS staging:
// chunk i+1 is issued before computing chunk i; s_wait_asynccnt 4 lets
// the next chunk stay in flight while WMMAs run on the current one.
// =====================================================================
__global__ __launch_bounds__(256) void fc1_wmma(
    const unsigned short* __restrict__ Abf,  // h2bf [1024][32768]
    const unsigned short* __restrict__ Bbf,  // fw1bf [1024][32768]
    const float* __restrict__ bias,          // [1024]
    float* __restrict__ Cout)                // [1024][1024]
{
    __shared__ __align__(16) unsigned int lsA[2][128][16];  // 2 x 8 KB
    __shared__ __align__(16) unsigned int lsB[2][128][16];  // 2 x 8 KB
    const int tid = threadIdx.x;
    const int wave = tid >> 5, lane = tid & 31;
    const int m0 = blockIdx.x * 128;
    const int n0 = blockIdx.y * 128;

    v8f acc[8];
#pragma unroll
    for (int nt = 0; nt < 8; ++nt)
#pragma unroll
        for (int j = 0; j < 8; ++j) acc[nt][j] = 0.f;

    const int r  = tid >> 1;
    const int cb = (tid & 1) * 8;
    const unsigned gArow = (unsigned)((m0 + r) * 65536);  // row byte offset (stride 32768*2)
    const unsigned gBrow = (unsigned)((n0 + r) * 65536);
    const unsigned ldsAme = lds_off32(&lsA[0][0][0]) + (unsigned)((r * 16 + cb) * 4);
    const unsigned ldsBme = lds_off32(&lsB[0][0][0]) + (unsigned)((r * 16 + cb) * 4);

    auto issue = [&](int kk, int bufsel) {
        const unsigned kb = (unsigned)(kk * 2 + cb * 4);
        const unsigned so = (unsigned)bufsel * 8192u;
        async_ld_b128(ldsAme + so,      gArow + kb,      (const void*)Abf);
        async_ld_b128(ldsAme + so + 16, gArow + kb + 16, (const void*)Abf);
        async_ld_b128(ldsBme + so,      gBrow + kb,      (const void*)Bbf);
        async_ld_b128(ldsBme + so + 16, gBrow + kb + 16, (const void*)Bbf);
    };

    issue(0, 0);                                   // prologue: chunk 0 -> buf 0
    int cur = 0;
    for (int k0 = 0; k0 < 32768; k0 += 32) {
        const int kn = (k0 + 32 < 32768) ? (k0 + 32) : 0;   // wrap: harmless dummy
        issue(kn, cur ^ 1);                        // prefetch next chunk
        wait_async4();                             // current chunk resident; next in flight
        __syncthreads();

        const int am = wave * 16 + (lane & 15);
        const int a0 = (lane >= 16) ? 1 : 0;
        const v4u* pA = (const v4u*)&lsA[cur][am][0];
        FragU af;
        af.u4[0] = pA[a0];
        af.u4[1] = pA[a0 + 2];
        const int i0 = (lane >= 16) ? 1 : 0;
#pragma unroll
        for (int nt = 0; nt < 8; ++nt) {
            const int bn = nt * 16 + (lane & 15);
            const v4u* pB = (const v4u*)&lsB[cur][bn][0];
            FragU bfr;
            bfr.u4[0] = pB[i0];
            bfr.u4[1] = pB[i0 + 2];
            acc[nt] = __builtin_amdgcn_wmma_f32_16x16x32_bf16(
                false, af.bf, false, bfr.bf, (short)0, acc[nt], false, false);
        }
        __syncthreads();                           // readers done before buf reuse
        cur ^= 1;
    }
    wait_async0();                                 // drain dummy prefetch

    const int mrow0 = m0 + wave * 16 + ((lane >= 16) ? 8 : 0);
#pragma unroll
    for (int nt = 0; nt < 8; ++nt) {
        const int n = n0 + nt * 16 + (lane & 15);
        const float bv = bias[n];
#pragma unroll
        for (int i = 0; i < 8; ++i) {
            float v = acc[nt][i] + bv;
            Cout[(size_t)(mrow0 + i) * 1024 + n] = v > 0.f ? v : 0.f;
        }
    }
}

// ---------- small helpers ----------
__global__ void cvt_bf16(const float* __restrict__ src, unsigned short* __restrict__ dst, int n) {
    int i = blockIdx.x * blockDim.x + threadIdx.x;
    int stride = gridDim.x * blockDim.x;
    for (; i < n; i += stride) dst[i] = bfbits(src[i]);
}

__global__ void cvt_w2pad(const float* __restrict__ w2, unsigned short* __restrict__ dst) {
    int i = blockIdx.x * blockDim.x + threadIdx.x;     // 128*608
    if (i >= 128 * 608) return;
    int m = i / 608, rr = i - m * 608;
    float v = (rr < 594) ? w2[m * 594 + rr] : 0.f;
    dst[i] = bfbits(v);
}

__global__ void fc2_kernel(const float* __restrict__ h, const float* __restrict__ fw2,
                           const float* __restrict__ fb2, float* __restrict__ out) {
    int i = blockIdx.x * blockDim.x + threadIdx.x;
    if (i >= 1024 * 10) return;
    int m = i / 10, n = i - m * 10;
    const float* hr = h + (size_t)m * 1024;
    const float* wr = fw2 + (size_t)n * 1024;
    float acc = fb2[n];
    for (int k = 0; k < 1024; ++k) acc += hr[k] * wr[k];
    out[i] = acc;
}

// =====================================================================
extern "C" void kernel_launch(void* const* d_in, const int* in_sizes, int n_in,
                              void* d_out, int out_size, void* d_ws, size_t ws_size,
                              hipStream_t stream) {
    (void)in_sizes; (void)n_in; (void)out_size; (void)ws_size;
    const float* x   = (const float*)d_in[0];
    const float* w1  = (const float*)d_in[1];
    const float* b1  = (const float*)d_in[2];
    const float* w2  = (const float*)d_in[3];
    const float* b2  = (const float*)d_in[4];
    const float* fw1 = (const float*)d_in[5];
    const float* fb1 = (const float*)d_in[6];
    const float* fw2 = (const float*)d_in[7];
    const float* fb2 = (const float*)d_in[8];
    float* out = (float*)d_out;

    char* ws = (char*)d_ws;
    float*          h1    = (float*)ws;          ws += (size_t)1024 * 64 * 256 * 4;   // 64 MB
    unsigned short* h2bf  = (unsigned short*)ws; ws += (size_t)1024 * 32768 * 2;      // 64 MB
    unsigned short* fw1bf = (unsigned short*)ws; ws += (size_t)1024 * 32768 * 2;      // 64 MB
    unsigned short* w2bf  = (unsigned short*)ws; ws += (size_t)128 * 608 * 2;
    float*          fc1o  = (float*)ws;                                                // 4 MB

    cvt_bf16<<<2048, 256, 0, stream>>>(fw1, fw1bf, 1024 * 32768);
    cvt_w2pad<<<(128 * 608 + 255) / 256, 256, 0, stream>>>(w2, w2bf);
    layer1_knn<<<1024, 256, 0, stream>>>(x, w1, b1, h1);
    layer2_knn_wmma<<<1024, 256, 0, stream>>>(h1, w2bf, b2, h2bf);
    dim3 g(8, 8);
    fc1_wmma<<<g, 256, 0, stream>>>(h2bf, fw1bf, fb1, fc1o);
    fc2_kernel<<<(1024 * 10 + 255) / 256, 256, 0, stream>>>(fc1o, fw2, fb2, out);
}
